// GraphTransformer_16329465660181
// MI455X (gfx1250) — matrix operand order repaired and checked by reading, not compiled, not run on hardware
//
#include <hip/hip_runtime.h>
#include <hip/hip_bf16.h>

// ---------------------------------------------------------------------------
// Graph Transformer for MI455X (gfx1250, wave32, WMMA).
//  - All dense GEMMs use v_wmma_f32_16x16x32_f16 (f16 in, f32 accumulate).
//  - Each wave computes a 16x64 D tile (4 WMMA acc) -> A-fragment reuse x4.
//  - Segment softmax / segment sums via L2 atomics (bandwidth-bound part).
// ---------------------------------------------------------------------------

#define NN   100000   // nodes  (== 6250 * 16, no padding needed)
#define EE   640000   // edges
#define FIN  256
#define HD   128
#define NH   8        // heads
#define DH   16       // per-head dim
#define NL   4
#define NG   256

typedef __attribute__((ext_vector_type(16))) _Float16 v16h;
typedef __attribute__((ext_vector_type(8)))  float    v8f;

union Frag16 { uint4 u[2]; v16h h; };

// ---- order-preserving float <-> uint encoding for atomic max -------------
__device__ __forceinline__ unsigned int enc_f(float f) {
    unsigned int b = __float_as_uint(f);
    return (b & 0x80000000u) ? ~b : (b | 0x80000000u);
}
__device__ __forceinline__ float dec_f(unsigned int u) {
    if (u == 0u) return 0.0f;                       // "never touched" sentinel
    unsigned int b = (u & 0x80000000u) ? (u ^ 0x80000000u) : ~u;
    return __uint_as_float(b);
}

__device__ __forceinline__ unsigned int pack_h2(float lo, float hi) {
    union { _Float16 h; unsigned short u; } a, b;
    a.h = (_Float16)lo; b.h = (_Float16)hi;
    return (unsigned int)a.u | ((unsigned int)b.u << 16);
}

// ---------------------------------------------------------------------------
// f32 -> f16 convert
// ---------------------------------------------------------------------------
__global__ void k_f32_to_f16(const float* __restrict__ in, _Float16* __restrict__ out,
                             long long n) {
    long long i = (long long)blockIdx.x * blockDim.x + threadIdx.x;
    if (i < n) out[i] = (_Float16)in[i];
}

// ---------------------------------------------------------------------------
// Pack a row-major [K, Ntot] f32 weight matrix into WMMA B fragments (f16).
// Layout (16-bit B 32x16, contiguous k order, not the ISA table order): fragment index
//   frag[((nTile*kChunks + kc)*32 + lane)*8 + v]
//   col = nTile*16 + (lane&15);  kbase = kc*32 + (lane>>4)*16 + 2*v
// ---------------------------------------------------------------------------
__global__ void k_pack_b(const float* __restrict__ W, unsigned int* __restrict__ out,
                         int K, int Ntot) {
    int kChunks = K >> 5, nTiles = Ntot >> 4;
    int total = nTiles * kChunks * 32 * 8;
    int idx = blockIdx.x * blockDim.x + threadIdx.x;
    if (idx >= total) return;
    int v    = idx & 7;
    int lane = (idx >> 3) & 31;
    int rest = idx >> 8;
    int kc    = rest % kChunks;
    int nTile = rest / kChunks;
    int col   = nTile * 16 + (lane & 15);
    int kbase = kc * 32 + ((lane >> 4) << 4) + 2 * v;
    out[idx] = pack_h2(W[(size_t)kbase * Ntot + col],
                       W[(size_t)(kbase + 1) * Ntot + col]);
}

// Pack per-layer Wq|Wk|Wv|Ws (each [128,128]) into one [128,512] B-fragment set.
__global__ void k_pack_qkvs(const float* __restrict__ Wq, const float* __restrict__ Wk,
                            const float* __restrict__ Wv, const float* __restrict__ Wsk,
                            unsigned int* __restrict__ out, int layer) {
    const int total = 32 * 4 * 32 * 8;     // nTiles=32, kChunks=4
    int idx = blockIdx.x * blockDim.x + threadIdx.x;
    if (idx >= total) return;
    int v    = idx & 7;
    int lane = (idx >> 3) & 31;
    int rest = idx >> 8;
    int kc    = rest & 3;
    int nTile = rest >> 2;
    int col   = nTile * 16 + (lane & 15);
    int kbase = kc * 32 + ((lane >> 4) << 4) + 2 * v;
    const float* Wm = (col < 128) ? Wq : (col < 256) ? Wk : (col < 384) ? Wv : Wsk;
    Wm += (size_t)layer * HD * HD;
    int lc = col & 127;
    out[idx] = pack_h2(Wm[(size_t)kbase * HD + lc],
                       Wm[(size_t)(kbase + 1) * HD + lc]);
}

// Concatenate bq|bk|bv|bs -> [NL,512]
__global__ void k_pack_bias(const float* __restrict__ bq, const float* __restrict__ bk,
                            const float* __restrict__ bv, const float* __restrict__ bs,
                            float* __restrict__ out) {
    int t = blockIdx.x * blockDim.x + threadIdx.x;
    if (t >= NL * 512) return;
    int l = t >> 9, c = t & 511;
    const float* b = (c < 128) ? bq : (c < 256) ? bk : (c < 384) ? bv : bs;
    out[t] = b[l * HD + (c & 127)];
}

// ---------------------------------------------------------------------------
// WMMA GEMM:  C[M,Ntot] = A[M,K](f16, row-major) @ B(packed f16) + bias
// One wave computes a 16x64 D tile: 1 A fragment feeds 4 WMMAs per K-chunk.
// A fragment (16-bit A 16x32, contiguous k order, same as packed B): row = mTile*16+(lane&15),
//   lanes<16 hold K {0..15}, lanes>=16 hold K {16..31}
//   -> two contiguous 16B loads per lane.
// Requires Ntot % 64 == 0.
// ---------------------------------------------------------------------------
__global__ void k_wmma_gemm(const _Float16* __restrict__ A,
                            const unsigned int* __restrict__ Bp,
                            const float* __restrict__ bias,
                            float* __restrict__ Cf32,
                            _Float16* __restrict__ Cf16,
                            int M, int K, int Ntot) {
    int nGroups = Ntot >> 6;                 // 4 N-tiles (64 cols) per wave
    int mTiles  = M >> 4;
    int tile = blockIdx.x * (blockDim.x >> 5) + (threadIdx.x >> 5);
    if (tile >= mTiles * nGroups) return;
    int lane  = threadIdx.x & 31;
    int mTile = tile / nGroups;
    int nGrp  = tile % nGroups;
    int nT0   = nGrp * 4;                    // first of 4 consecutive N-tiles

    int row  = mTile * 16 + (lane & 15);
    int koff = (lane >> 4) * 16;
    int kChunks = K >> 5;

    v8f acc0 = {}, acc1 = {}, acc2 = {}, acc3 = {};
    const _Float16* arow = A + (size_t)row * K;
    // packed B stream for N-tile t: Bp + ((nTile*kChunks + kc)*32 + lane)*8
    const unsigned int* b0 = Bp + ((size_t)(nT0 + 0) * kChunks * 32 + lane) * 8;
    const unsigned int* b1 = Bp + ((size_t)(nT0 + 1) * kChunks * 32 + lane) * 8;
    const unsigned int* b2 = Bp + ((size_t)(nT0 + 2) * kChunks * 32 + lane) * 8;
    const unsigned int* b3 = Bp + ((size_t)(nT0 + 3) * kChunks * 32 + lane) * 8;

    for (int kc = 0; kc < kChunks; ++kc) {
        Frag16 a, f0, f1, f2, f3;
        int k0 = kc * 32;
        a.u[0] = *(const uint4*)(arow + k0 + koff);
        a.u[1] = *(const uint4*)(arow + k0 + koff + 8);
        size_t bo = (size_t)kc * 32 * 8;
        f0.u[0] = *(const uint4*)(b0 + bo);     f0.u[1] = *(const uint4*)(b0 + bo + 4);
        f1.u[0] = *(const uint4*)(b1 + bo);     f1.u[1] = *(const uint4*)(b1 + bo + 4);
        f2.u[0] = *(const uint4*)(b2 + bo);     f2.u[1] = *(const uint4*)(b2 + bo + 4);
        f3.u[0] = *(const uint4*)(b3 + bo);     f3.u[1] = *(const uint4*)(b3 + bo + 4);
        acc0 = __builtin_amdgcn_wmma_f32_16x16x32_f16(false, a.h, false, f0.h, (short)0, acc0, false, false);
        acc1 = __builtin_amdgcn_wmma_f32_16x16x32_f16(false, a.h, false, f1.h, (short)0, acc1, false, false);
        acc2 = __builtin_amdgcn_wmma_f32_16x16x32_f16(false, a.h, false, f2.h, (short)0, acc2, false, false);
        acc3 = __builtin_amdgcn_wmma_f32_16x16x32_f16(false, a.h, false, f3.h, (short)0, acc3, false, false);
    }

    // D layout: VGPR r -> (lanes 0-15: M=r, N=lane) (lanes 16-31: M=r+8, N=lane-16)
    int m0 = mTile * 16 + ((lane >> 4) ? 8 : 0);
    v8f accs[4] = {acc0, acc1, acc2, acc3};
#pragma unroll
    for (int t = 0; t < 4; ++t) {
        int col = (nT0 + t) * 16 + (lane & 15);
        float bv_ = bias[col];
#pragma unroll
        for (int r = 0; r < 8; ++r) {
            float val = accs[t][r] + bv_;
            size_t off = (size_t)(m0 + r) * Ntot + col;
            if (Cf32) Cf32[off] = val;
            if (Cf16) Cf16[off] = (_Float16)val;
        }
    }
}

// ---------------------------------------------------------------------------
// Edge phase 1: alpha[e,h] = (q[dst,h,:] . k[src,h,:]) / 4 ;  segment max
// qkvs layout per node: [q(128) | k(128) | v(128) | skip(128)]
// ---------------------------------------------------------------------------
__global__ void k_edge_alpha(const long long* __restrict__ ei,
                             const float* __restrict__ qkvs,
                             float* __restrict__ alpha,
                             unsigned int* __restrict__ menc) {
    long long gid = (long long)blockIdx.x * blockDim.x + threadIdx.x;
    if (gid >= (long long)EE * NH) return;
    int e = (int)(gid >> 3), h = (int)(gid & 7);
    int src = (int)ei[e];
    int dst = (int)ei[(long long)EE + e];
    const float4* qp = (const float4*)(qkvs + (size_t)dst * 512 + h * DH);
    const float4* kp = (const float4*)(qkvs + (size_t)src * 512 + 128 + h * DH);
    float d = 0.f;
#pragma unroll
    for (int r = 0; r < 4; ++r) {
        float4 a = qp[r], b = kp[r];
        d += a.x * b.x + a.y * b.y + a.z * b.z + a.w * b.w;
    }
    float al = d * 0.25f;                       // 1/sqrt(16)
    alpha[(size_t)e * NH + h] = al;
    atomicMax(&menc[(size_t)dst * NH + h], enc_f(al));
}

// Edge phase 2: e = exp(alpha - m[dst]); denom[dst] += e; agg[dst] += e * v[src]
__global__ void k_edge_accum(const long long* __restrict__ ei,
                             const float* __restrict__ qkvs,
                             const float* __restrict__ alpha,
                             const unsigned int* __restrict__ menc,
                             float* __restrict__ agg,
                             float* __restrict__ denom) {
    long long gid = (long long)blockIdx.x * blockDim.x + threadIdx.x;
    if (gid >= (long long)EE * HD) return;
    int e = (int)(gid >> 7), c = (int)(gid & 127), h = c >> 4;
    int src = (int)ei[e];
    int dst = (int)ei[(long long)EE + e];
    float a  = alpha[(size_t)e * NH + h];
    float m  = dec_f(menc[(size_t)dst * NH + h]);
    float ev = expf(a - m);
    float vv = qkvs[(size_t)src * 512 + 256 + c];
    atomicAdd(&agg[(size_t)dst * HD + c], ev * vv);
    if ((c & 15) == 0) atomicAdd(&denom[(size_t)dst * NH + h], ev);
}

// ---------------------------------------------------------------------------
// Node update: attn-out + skip -> LayerNorm -> exact GELU -> + residual
// One block of 128 threads per node.
// ---------------------------------------------------------------------------
__global__ void k_node_update(const float* __restrict__ agg,
                              const float* __restrict__ denom,
                              const float* __restrict__ qkvs,
                              const float* __restrict__ res,
                              const float* __restrict__ ln_w,
                              const float* __restrict__ ln_b,
                              float* __restrict__ hout,
                              _Float16* __restrict__ hout16,
                              int layer) {
    int n = blockIdx.x, c = threadIdx.x;
    float val = agg[(size_t)n * HD + c] / fmaxf(denom[(size_t)n * NH + (c >> 4)], 1e-16f)
              + qkvs[(size_t)n * 512 + 384 + c];          // skip (bias included)
    __shared__ float red[HD];
    red[c] = val; __syncthreads();
    for (int s = 64; s > 0; s >>= 1) { if (c < s) red[c] += red[c + s]; __syncthreads(); }
    float mu = red[0] * (1.0f / HD);
    __syncthreads();
    float d = val - mu;
    red[c] = d * d; __syncthreads();
    for (int s = 64; s > 0; s >>= 1) { if (c < s) red[c] += red[c + s]; __syncthreads(); }
    float var = red[0] * (1.0f / HD);
    float y = d * rsqrtf(var + 1e-5f) * ln_w[layer * HD + c] + ln_b[layer * HD + c];
    float g = 0.5f * y * (1.0f + erff(y * 0.70710678118654752f));   // exact GELU
    float o = g + res[(size_t)n * HD + c];
    hout[(size_t)n * HD + c] = o;
    hout16[(size_t)n * HD + c] = (_Float16)o;
}

// ---------------------------------------------------------------------------
// Graph pooling: sum / max(encoded) / count via atomics
// ---------------------------------------------------------------------------
__global__ void k_pool(const float* __restrict__ h, const long long* __restrict__ batch,
                       float* __restrict__ ssum, unsigned int* __restrict__ mxenc,
                       float* __restrict__ cnt) {
    long long gid = (long long)blockIdx.x * blockDim.x + threadIdx.x;
    if (gid >= (long long)NN * HD) return;
    int n = (int)(gid >> 7), c = (int)(gid & 127);
    int g = (int)batch[n];
    float v = h[gid];
    atomicAdd(&ssum[(size_t)g * HD + c], v);
    atomicMax(&mxenc[(size_t)g * HD + c], enc_f(v));
    if (c == 0) atomicAdd(&cnt[g], 1.0f);
}

// Final MLP: [G,384] -> relu(384x256) -> relu(256x128) -> 128x1. One block / graph.
__global__ void k_final_mlp(const float* __restrict__ ssum,
                            const unsigned int* __restrict__ mxenc,
                            const float* __restrict__ cnt,
                            const float* __restrict__ W1, const float* __restrict__ b1,
                            const float* __restrict__ W2, const float* __restrict__ b2,
                            const float* __restrict__ W3, const float* __restrict__ b3,
                            float* __restrict__ out) {
    int g = blockIdx.x, t = threadIdx.x;          // 256 threads
    __shared__ float gv[3 * HD];
    __shared__ float h1[2 * HD];
    __shared__ float h2[HD];
    float cn = fmaxf(cnt[g], 1.0f);
    if (t < HD) {
        float sv = ssum[(size_t)g * HD + t];
        gv[t]          = sv / cn;                 // mean
        gv[HD + t]     = dec_f(mxenc[(size_t)g * HD + t]);  // max (0 if empty)
        gv[2 * HD + t] = sv;                      // sum
    }
    __syncthreads();
    float a1 = b1[t];
    for (int j = 0; j < 3 * HD; ++j) a1 += gv[j] * W1[(size_t)j * 256 + t];
    h1[t] = fmaxf(a1, 0.0f);
    __syncthreads();
    if (t < HD) {
        float a2 = b2[t];
        for (int j = 0; j < 2 * HD; ++j) a2 += h1[j] * W2[(size_t)j * HD + t];
        h2[t] = fmaxf(a2, 0.0f) * W3[t];
    }
    __syncthreads();
    for (int s = 64; s > 0; s >>= 1) { if (t < s) h2[t] += h2[t + s]; __syncthreads(); }
    if (t == 0) out[g] = h2[0] + b3[0];
}

// ---------------------------------------------------------------------------
// Host-side launch
// ---------------------------------------------------------------------------
extern "C" void kernel_launch(void* const* d_in, const int* in_sizes, int n_in,
                              void* d_out, int out_size, void* d_ws, size_t ws_size,
                              hipStream_t stream) {
    (void)in_sizes; (void)n_in; (void)out_size; (void)ws_size;

    const float*     x    = (const float*)d_in[0];
    const long long* ei   = (const long long*)d_in[1];
    const long long* bat  = (const long long*)d_in[2];
    const float* W_in = (const float*)d_in[3];
    const float* b_in = (const float*)d_in[4];
    const float* Wq = (const float*)d_in[5];
    const float* bq = (const float*)d_in[6];
    const float* Wk = (const float*)d_in[7];
    const float* bk = (const float*)d_in[8];
    const float* Wv = (const float*)d_in[9];
    const float* bv = (const float*)d_in[10];
    const float* Ws = (const float*)d_in[11];
    const float* bs = (const float*)d_in[12];
    const float* ln_w = (const float*)d_in[13];
    const float* ln_b = (const float*)d_in[14];
    const float* W1 = (const float*)d_in[15];
    const float* b1 = (const float*)d_in[16];
    const float* W2 = (const float*)d_in[17];
    const float* b2 = (const float*)d_in[18];
    const float* W3 = (const float*)d_in[19];
    const float* b3 = (const float*)d_in[20];
    float* out = (float*)d_out;

    // ---- workspace carve-out (deterministic every call) ----
    char* w = (char*)d_ws;
    auto alloc = [&](size_t bytes) -> char* {
        char* p = w; w += (bytes + 255) & ~(size_t)255; return p;
    };
    _Float16* xf16  = (_Float16*)alloc((size_t)NN * FIN * 2);
    _Float16* hf16  = (_Float16*)alloc((size_t)NN * HD * 2);
    float*    hA    = (float*)   alloc((size_t)NN * HD * 4);
    float*    hB    = (float*)   alloc((size_t)NN * HD * 4);
    float*    qkvs  = (float*)   alloc((size_t)NN * 512 * 4);
    float*    agg   = (float*)   alloc((size_t)NN * HD * 4);
    float*    denom = (float*)   alloc((size_t)NN * NH * 4);
    unsigned int* menc = (unsigned int*)alloc((size_t)NN * NH * 4);
    float*    alpha = (float*)   alloc((size_t)EE * NH * 4);
    unsigned int* pWin  = (unsigned int*)alloc((size_t)(FIN/32)*(HD/16)*32*8*4);
    unsigned int* pQKVS = (unsigned int*)alloc((size_t)NL * (HD/32)*(512/16)*32*8*4);
    float*    biasQ = (float*)alloc((size_t)NL * 512 * 4);
    float*    pcnt  = (float*)alloc((size_t)NG * 4);
    float*    psum  = (float*)alloc((size_t)NG * HD * 4);
    unsigned int* pmax = (unsigned int*)alloc((size_t)NG * HD * 4);

    const int TB = 256;

    // 0) convert x -> f16; pack weights
    {
        long long nx = (long long)NN * FIN;
        k_f32_to_f16<<<(unsigned)((nx + TB - 1) / TB), TB, 0, stream>>>(x, xf16, nx);
        int totWin = (FIN/32)*(HD/16)*32*8;
        k_pack_b<<<(totWin + TB - 1)/TB, TB, 0, stream>>>(W_in, pWin, FIN, HD);
        const int totL = (HD/32)*(512/16)*32*8;   // 32768 per layer
        for (int l = 0; l < NL; ++l)
            k_pack_qkvs<<<(totL + TB - 1)/TB, TB, 0, stream>>>(Wq, Wk, Wv, Ws,
                                                               pQKVS + (size_t)l*totL, l);
        k_pack_bias<<<(NL*512 + TB - 1)/TB, TB, 0, stream>>>(bq, bk, bv, bs, biasQ);
    }

    // 1) h = x @ W_in + b_in      (WMMA, K=256, Ntot=128, 16x64 tiles)
    {
        int tiles = (NN/16) * (HD/64);            // 12500 waves
        k_wmma_gemm<<<(tiles + 7)/8, 256, 0, stream>>>(xf16, pWin, b_in, hA, hf16,
                                                       NN, FIN, HD);
    }

    float* hcur = hA;
    float* hnxt = hB;

    // 2) 4 transformer layers
    for (int l = 0; l < NL; ++l) {
        hipMemsetAsync(agg,   0, (size_t)NN * HD * 4, stream);
        hipMemsetAsync(denom, 0, (size_t)NN * NH * 4, stream);
        hipMemsetAsync(menc,  0, (size_t)NN * NH * 4, stream);

        // qkvs = h @ [Wq|Wk|Wv|Ws] + bias      (WMMA, K=128, Ntot=512)
        int tiles = (NN/16) * (512/64);           // 50000 waves
        k_wmma_gemm<<<(tiles + 7)/8, 256, 0, stream>>>(hf16, pQKVS + (size_t)l*32768,
                                                       biasQ + (size_t)l*512,
                                                       qkvs, (_Float16*)nullptr,
                                                       NN, HD, 512);

        long long ne1 = (long long)EE * NH;
        k_edge_alpha<<<(unsigned)((ne1 + TB - 1)/TB), TB, 0, stream>>>(ei, qkvs, alpha, menc);
        long long ne2 = (long long)EE * HD;
        k_edge_accum<<<(unsigned)((ne2 + TB - 1)/TB), TB, 0, stream>>>(ei, qkvs, alpha, menc,
                                                                       agg, denom);
        k_node_update<<<NN, HD, 0, stream>>>(agg, denom, qkvs, hcur, ln_w, ln_b,
                                             hnxt, hf16, l);
        float* t = hcur; hcur = hnxt; hnxt = t;
    }

    // 3) pooling + final MLP
    hipMemsetAsync(pcnt, 0, (size_t)NG * 4, stream);
    hipMemsetAsync(psum, 0, (size_t)NG * HD * 4, stream);
    hipMemsetAsync(pmax, 0, (size_t)NG * HD * 4, stream);
    {
        long long np = (long long)NN * HD;
        k_pool<<<(unsigned)((np + TB - 1)/TB), TB, 0, stream>>>(hcur, bat, psum, pmax, pcnt);
        k_final_mlp<<<NG, 256, 0, stream>>>(psum, pmax, pcnt, W1, b1, W2, b2, W3, b3, out);
    }
}
